// CrossAttention_27590869909610
// MI455X (gfx1250) — matrix operand order repaired and checked by reading, not compiled, not run on hardware
//
#include <hip/hip_runtime.h>
#include <hip/hip_bf16.h>

// ---- problem constants (match reference) ----
#define HEADS  8
#define DH     64
#define INNER  512      // HEADS*DH
#define BATCH  4
#define NQ     2048
#define NC     2048
#define DQ     1024
#define DC     768
#define QSCALE 0.125f   // 64^-0.5

typedef __attribute__((ext_vector_type(16))) __bf16 v16bf;
typedef __attribute__((ext_vector_type(8)))  __bf16 v8bf;
typedef __attribute__((ext_vector_type(8)))  float  v8f;
typedef __attribute__((ext_vector_type(4)))  unsigned v4u;
typedef __attribute__((ext_vector_type(8)))  int   v8i;
typedef __attribute__((ext_vector_type(4)))  int   v4i;

#if defined(__HIP_DEVICE_COMPILE__) && __has_builtin(__builtin_amdgcn_tensor_load_to_lds) && \
    __has_builtin(__builtin_amdgcn_s_wait_tensorcnt)
#define USE_TDM 1
#else
#define USE_TDM 0
#endif

// ---------------------------------------------------------------------------
// WMMA fragment helpers (CDNA5 16x16x32 bf16, wave32).
// A operand: lane (m = lane&15, g = lane>>4) holds row m, K = {8g..8g+7} U {16+8g..16+8g+7}
// B operand: lane (n = lane&15, g = lane>>4) holds col n, K = {8g..8g+7} U {16+8g..16+8g+7}
// C/D:       VGPR r <-> M = r + 8*(lane>>4), N = lane&15
// ---------------------------------------------------------------------------
__device__ __forceinline__ v16bf load_a_frag(const __bf16* base, int ld) {
  const int lane = threadIdx.x & 31;
  const int m = lane & 15, g = lane >> 4;
  const __bf16* p = base + m * ld + 8 * g;
  v8bf lo = *(const v8bf*)(p);
  v8bf hi = *(const v8bf*)(p + 16);
  return __builtin_shufflevector(lo, hi, 0,1,2,3,4,5,6,7,8,9,10,11,12,13,14,15);
}

__device__ __forceinline__ v16bf load_b_frag(const __bf16* base, int ld) {
  const int lane = threadIdx.x & 31;
  const int n = lane & 15, g = lane >> 4;
  const __bf16* p = base + n * ld + 8 * g;
  v8bf lo = *(const v8bf*)(p);
  v8bf hi = *(const v8bf*)(p + 16);
  return __builtin_shufflevector(lo, hi, 0,1,2,3,4,5,6,7,8,9,10,11,12,13,14,15);
}

__device__ __forceinline__ v8f wmma_bf16(v16bf a, v16bf b, v8f c) {
  return __builtin_amdgcn_wmma_f32_16x16x32_bf16(
      /*neg_a=*/false, a, /*neg_b=*/false, b,
      /*c_mod=*/(short)0, c, /*reuse_a=*/false, /*reuse_b=*/false);
}

#if USE_TDM
// TDM load of a 2D bf16 tile: tile_h rows x tile_w elements, row pitch
// `stride_elems` (elements), from `gptr` into LDS byte offset `lds_addr`.
// D# layout per CDNA5 ISA (data_size=1 -> 2B).
__device__ __forceinline__ void tdm_load_2d(unsigned lds_addr, const __bf16* gptr,
                                            unsigned tile_w, unsigned tile_h,
                                            unsigned stride_elems) {
  unsigned long long ga = (unsigned long long)gptr;
  v4u g0;
  g0[0] = 1u;                                      // count=1 (valid), user mode
  g0[1] = lds_addr;                                // lds_addr (bytes)
  g0[2] = (unsigned)ga;                            // global_addr[31:0]
  g0[3] = (unsigned)((ga >> 32) & 0x01FFFFFFu) |   // global_addr[56:32]
          0x80000000u;                             // type=2 ("image")
  v8i g1;
  g1[0] = 0x00010000;                              // wg_mask=0, data_size=1 (2B)
  g1[1] = (int)(tile_w << 16);                     // tensor_dim0[15:0] @ bits 63:48
  g1[2] = (int)(tile_h << 16);                     // dim0 hi=0, tensor_dim1[15:0]
  g1[3] = (int)(tile_w << 16);                     // dim1 hi=0, tile_dim0
  g1[4] = (int)tile_h;                             // tile_dim1, tile_dim2=0
  g1[5] = (int)stride_elems;                       // tensor_dim0_stride[31:0]
  g1[6] = 0;                                       // stride[47:32], dim1_stride lo
  g1[7] = 0;                                       // dim1_stride hi
  v4i z4 = {0, 0, 0, 0};
#if __clang_major__ >= 23
  v8i z8 = {0, 0, 0, 0, 0, 0, 0, 0};
  __builtin_amdgcn_tensor_load_to_lds(g0, g1, z4, z4, z8, 0);
#else
  __builtin_amdgcn_tensor_load_to_lds(g0, g1, z4, z4, 0);
#endif
}
#endif

// ---------------------------------------------------------------------------
// Conversion kernels
// ---------------------------------------------------------------------------
__global__ __launch_bounds__(256) void cvt_f32_bf16(const float* __restrict__ in,
                                                    __bf16* __restrict__ out, size_t n) {
  size_t i = (size_t)blockIdx.x * 256 + threadIdx.x;
  if (i < n) out[i] = (__bf16)in[i];
}

// out[c][r] = in[r][c]  (in: R x C f32, out: C x R bf16)
__global__ __launch_bounds__(256) void trans_f32_bf16(const float* __restrict__ in,
                                                      __bf16* __restrict__ out, int R, int C) {
  size_t i = (size_t)blockIdx.x * 256 + threadIdx.x;
  if (i < (size_t)R * C) {
    int r = (int)(i / C), c = (int)(i % C);
    out[(size_t)c * R + r] = (__bf16)in[i];
  }
}

// ---------------------------------------------------------------------------
// GEMM: C[M][N] = alpha * A[M][K] * Bt[N][K]^T  (A,Bt bf16; f32 accumulate)
// Block tile 128x128, BK=32, 256 threads = 8 waves (grid 4Mx2N), wave tile 32x64.
// LDS tiles double-buffered and streamed by the Tensor Data Mover: wave 0
// issues next-tile TDM, then s_wait_tensorcnt(2) (in-order completion =>
// current tile landed) before the publishing barrier, so DMA overlaps WMMA.
// MODE 0: bf16 store C[m][n]              (q/k projections; alpha = scale)
// MODE 1: bf16 store transposed into vT: vT[(b*N + n)*batchRows + j], m = b*batchRows + j
// MODE 2: f32 store C[m][n] + bias[n]     (final output projection)
// ---------------------------------------------------------------------------
template <int MODE>
__global__ __launch_bounds__(256) void gemm_bt_kernel(
    const __bf16* __restrict__ A, const __bf16* __restrict__ Bt,
    void* __restrict__ Cout, int M, int N, int K,
    int lda, int ldb, int ldc, float alpha,
    const float* __restrict__ bias, int batchRows) {
  (void)M;
  __shared__ __bf16 aT[2][128 * 32];
  __shared__ __bf16 bT[2][128 * 32];
  const int tid  = threadIdx.x;
  const int lane = tid & 31, w = tid >> 5;
  const int wm = w >> 1, wn = w & 1;           // 4 x 2 wave grid
  const int bm = blockIdx.y * 128, bn = blockIdx.x * 128;
  const int half = lane >> 4, ln = lane & 15;

  v8f acc[2][4];
#pragma unroll
  for (int mi = 0; mi < 2; ++mi)
#pragma unroll
    for (int ni = 0; ni < 4; ++ni)
#pragma unroll
      for (int r = 0; r < 8; ++r) acc[mi][ni][r] = 0.f;

  const int nk = K / 32;
#if USE_TDM
  // Generic shared pointers carry the LDS byte offset in their low 32 bits.
  unsigned aT_lds[2], bT_lds[2];
#pragma unroll
  for (int i = 0; i < 2; ++i) {
    aT_lds[i] = (unsigned)(unsigned long long)(&aT[i][0]);
    bT_lds[i] = (unsigned)(unsigned long long)(&bT[i][0]);
  }
  if (w == 0) {  // one wave drives the TDM (EXEC is ignored by tensor ops)
    tdm_load_2d(aT_lds[0], A + (size_t)bm * lda, 32, 128, lda);
    tdm_load_2d(bT_lds[0], Bt + (size_t)bn * ldb, 32, 128, ldb);
  }
#endif

  for (int t = 0; t < nk; ++t) {
    const int cur = t & 1;
    __syncthreads();  // all waves done reading the spare buffer
#if USE_TDM
    if (w == 0) {
      if (t + 1 < nk) {
        tdm_load_2d(aT_lds[cur ^ 1], A + (size_t)bm * lda + (t + 1) * 32, 32, 128, lda);
        tdm_load_2d(bT_lds[cur ^ 1], Bt + (size_t)bn * ldb + (t + 1) * 32, 32, 128, ldb);
        __builtin_amdgcn_s_wait_tensorcnt(2);  // current tile complete
      } else {
        __builtin_amdgcn_s_wait_tensorcnt(0);
      }
    }
    __syncthreads();  // publish current tile
#else
#pragma unroll
    for (int p = 0; p < 2; ++p) {
      int idx = tid + p * 256;
      int row = idx >> 2, c8 = (idx & 3) * 8;
      *(v8bf*)(&aT[cur][row * 32 + c8]) = *(const v8bf*)(A + (size_t)(bm + row) * lda + t * 32 + c8);
      *(v8bf*)(&bT[cur][row * 32 + c8]) = *(const v8bf*)(Bt + (size_t)(bn + row) * ldb + t * 32 + c8);
    }
    __syncthreads();
#endif

    v16bf af[2], bfr[4];
#pragma unroll
    for (int mi = 0; mi < 2; ++mi) af[mi] = load_a_frag(&aT[cur][(wm * 32 + mi * 16) * 32], 32);
#pragma unroll
    for (int ni = 0; ni < 4; ++ni) bfr[ni] = load_b_frag(&bT[cur][(wn * 64 + ni * 16) * 32], 32);
#pragma unroll
    for (int mi = 0; mi < 2; ++mi)
#pragma unroll
      for (int ni = 0; ni < 4; ++ni)
        acc[mi][ni] = wmma_bf16(af[mi], bfr[ni], acc[mi][ni]);
  }

#pragma unroll
  for (int mi = 0; mi < 2; ++mi) {
    const int gm0 = bm + wm * 32 + mi * 16 + half * 8;
#pragma unroll
    for (int ni = 0; ni < 4; ++ni) {
      const int gn = bn + wn * 64 + ni * 16 + ln;
      if (MODE == 0) {
        __bf16* C = (__bf16*)Cout;
#pragma unroll
        for (int r = 0; r < 8; ++r)
          C[(size_t)(gm0 + r) * ldc + gn] = (__bf16)(acc[mi][ni][r] * alpha);
      } else if (MODE == 1) {
        __bf16* C = (__bf16*)Cout;
        const int bidx = gm0 / batchRows, j0 = gm0 % batchRows;
        v8bf pk;
#pragma unroll
        for (int r = 0; r < 8; ++r) pk[r] = (__bf16)(acc[mi][ni][r]);
        *(v8bf*)(C + ((size_t)bidx * N + gn) * batchRows + j0) = pk;
      } else {
        float* C = (float*)Cout;
        const float bb = bias[gn];
#pragma unroll
        for (int r = 0; r < 8; ++r)
          C[(size_t)(gm0 + r) * ldc + gn] = acc[mi][ni][r] + bb;
      }
    }
  }
}

// ---------------------------------------------------------------------------
// Flash attention: q,k bf16 [B*N][INNER] (scale pre-folded into q), vT bf16
// [B][INNER][NC]. Block = 256 thr = 8 waves; each wave owns 32 queries of a
// 256-query tile for one (b,h); 64-key chunks double-buffered in LDS and
// streamed by the TDM one chunk ahead of compute.
// ---------------------------------------------------------------------------
__global__ __launch_bounds__(256) void flash_attn_kernel(
    const __bf16* __restrict__ q, const __bf16* __restrict__ k,
    const __bf16* __restrict__ vT, __bf16* __restrict__ o) {
  __shared__ __bf16 kT[2][64 * 64];     // [key][d]
  __shared__ __bf16 vTile[2][64 * 64];  // [d][key]
  __shared__ __bf16 pS[8 * 32 * 64];    // per-wave P strips [m][key]

  const int tid = threadIdx.x, lane = tid & 31, w = tid >> 5;
  const int half = lane >> 4, ln = lane & 15;
  const int b = blockIdx.y / HEADS, h = blockIdx.y % HEADS;
  const int m0 = blockIdx.x * 256 + w * 32;

  const __bf16* qb = q + (size_t)b * NQ * INNER + h * DH;
  const __bf16* kb = k + (size_t)b * NC * INNER + h * DH;
  const __bf16* vb = vT + ((size_t)b * INNER + h * DH) * NC;
  __bf16* ob = o + (size_t)b * NQ * INNER + h * DH;
  __bf16* pw = &pS[w * 32 * 64];

  v16bf qa[2][2];
#pragma unroll
  for (int mi = 0; mi < 2; ++mi)
#pragma unroll
    for (int ks = 0; ks < 2; ++ks)
      qa[mi][ks] = load_a_frag(qb + (size_t)(m0 + mi * 16) * INNER + ks * 32, INNER);

  v8f om[2][4], rm[2], rs[2];
#pragma unroll
  for (int mi = 0; mi < 2; ++mi) {
#pragma unroll
    for (int r = 0; r < 8; ++r) { rm[mi][r] = -1e30f; rs[mi][r] = 0.f; }
#pragma unroll
    for (int ni = 0; ni < 4; ++ni)
#pragma unroll
      for (int r = 0; r < 8; ++r) om[mi][ni][r] = 0.f;
  }

#if USE_TDM
  unsigned kT_lds[2], vT_lds[2];
#pragma unroll
  for (int i = 0; i < 2; ++i) {
    kT_lds[i] = (unsigned)(unsigned long long)(&kT[i][0]);
    vT_lds[i] = (unsigned)(unsigned long long)(&vTile[i][0]);
  }
  if (w == 0) {  // prologue: stream chunk 0
    tdm_load_2d(kT_lds[0], kb, 64, 64, INNER);
    tdm_load_2d(vT_lds[0], vb, 64, 64, NC);
  }
#endif

  for (int j0 = 0, it = 0; j0 < NC; j0 += 64, ++it) {
    const int cur = it & 1;
    __syncthreads();  // spare buffer free (prior iteration's reads done)
#if USE_TDM
    if (w == 0) {
      if (j0 + 64 < NC) {
        tdm_load_2d(kT_lds[cur ^ 1], kb + (size_t)(j0 + 64) * INNER, 64, 64, INNER);
        tdm_load_2d(vT_lds[cur ^ 1], vb + (j0 + 64), 64, 64, NC);
        __builtin_amdgcn_s_wait_tensorcnt(2);  // chunk j0 landed (in-order)
      } else {
        __builtin_amdgcn_s_wait_tensorcnt(0);
      }
    }
    __syncthreads();  // publish chunk j0
#else
#pragma unroll
    for (int p = 0; p < 2; ++p) {
      int idx = tid + p * 256;
      int row = idx >> 3, c8 = (idx & 7) * 8;
      *(v8bf*)(&kT[cur][row * 64 + c8])    = *(const v8bf*)(kb + (size_t)(j0 + row) * INNER + c8);
      *(v8bf*)(&vTile[cur][row * 64 + c8]) = *(const v8bf*)(vb + (size_t)row * NC + j0 + c8);
    }
    __syncthreads();
    if (j0 + 64 < NC) {
      __builtin_prefetch(kb + (size_t)(j0 + 64) * INNER, 0, 1);
      __builtin_prefetch(vb + j0 + 64, 0, 1);
    }
#endif

    // S = Q * K^T  (B operand rows = key index -> kT is already Bt layout)
    v8f s[2][4];
#pragma unroll
    for (int mi = 0; mi < 2; ++mi)
#pragma unroll
      for (int ni = 0; ni < 4; ++ni)
#pragma unroll
        for (int r = 0; r < 8; ++r) s[mi][ni][r] = 0.f;
#pragma unroll
    for (int ks = 0; ks < 2; ++ks) {
      v16bf kf[4];
#pragma unroll
      for (int ni = 0; ni < 4; ++ni) kf[ni] = load_b_frag(&kT[cur][(ni * 16) * 64 + ks * 32], 64);
#pragma unroll
      for (int mi = 0; mi < 2; ++mi)
#pragma unroll
        for (int ni = 0; ni < 4; ++ni)
          s[mi][ni] = wmma_bf16(qa[mi][ks], kf[ni], s[mi][ni]);
    }

    // Online softmax (rows of S live across 16-lane groups; reduce via shfl_xor)
#pragma unroll
    for (int mi = 0; mi < 2; ++mi) {
      v8f cm;
#pragma unroll
      for (int r = 0; r < 8; ++r) {
        float m_ = fmaxf(fmaxf(s[mi][0][r], s[mi][1][r]), fmaxf(s[mi][2][r], s[mi][3][r]));
        cm[r] = m_;
      }
#pragma unroll
      for (int mask = 1; mask < 16; mask <<= 1)
#pragma unroll
        for (int r = 0; r < 8; ++r) cm[r] = fmaxf(cm[r], __shfl_xor(cm[r], mask, 32));

      v8f nm, al, ls;
#pragma unroll
      for (int r = 0; r < 8; ++r) {
        nm[r] = fmaxf(rm[mi][r], cm[r]);
        al[r] = __expf(rm[mi][r] - nm[r]);
        ls[r] = 0.f;
      }
      rm[mi] = nm;
#pragma unroll
      for (int ni = 0; ni < 4; ++ni)
#pragma unroll
        for (int r = 0; r < 8; ++r) {
          float e = __expf(s[mi][ni][r] - nm[r]);
          s[mi][ni][r] = e;
          ls[r] += e;
        }
#pragma unroll
      for (int mask = 1; mask < 16; mask <<= 1)
#pragma unroll
        for (int r = 0; r < 8; ++r) ls[r] += __shfl_xor(ls[r], mask, 32);
#pragma unroll
      for (int r = 0; r < 8; ++r) rs[mi][r] = rs[mi][r] * al[r] + ls[r];
#pragma unroll
      for (int ni = 0; ni < 4; ++ni)
#pragma unroll
        for (int r = 0; r < 8; ++r) om[mi][ni][r] *= al[r];

      // Re-layout P (C-frag -> LDS [m][key]); same-wave DS ops are in-order.
#pragma unroll
      for (int ni = 0; ni < 4; ++ni)
#pragma unroll
        for (int r = 0; r < 8; ++r)
          pw[(mi * 16 + half * 8 + r) * 64 + ni * 16 + ln] = (__bf16)s[mi][ni][r];
    }

    // O += P * V  (B operand rows = d -> vTile = V^T is Bt layout)
#pragma unroll
    for (int ks = 0; ks < 2; ++ks) {
      v16bf pa[2], vf[4];
#pragma unroll
      for (int mi = 0; mi < 2; ++mi) pa[mi] = load_a_frag(&pw[(mi * 16) * 64 + ks * 32], 64);
#pragma unroll
      for (int ni = 0; ni < 4; ++ni) vf[ni] = load_b_frag(&vTile[cur][(ni * 16) * 64 + ks * 32], 64);
#pragma unroll
      for (int mi = 0; mi < 2; ++mi)
#pragma unroll
        for (int ni = 0; ni < 4; ++ni)
          om[mi][ni] = wmma_bf16(pa[mi], vf[ni], om[mi][ni]);
    }
  }

  // Normalize and store attention output (bf16, feeds final GEMM)
#pragma unroll
  for (int mi = 0; mi < 2; ++mi)
#pragma unroll
    for (int ni = 0; ni < 4; ++ni)
#pragma unroll
      for (int r = 0; r < 8; ++r) {
        float v_ = om[mi][ni][r] / rs[mi][r];
        ob[(size_t)(m0 + mi * 16 + half * 8 + r) * INNER + ni * 16 + ln] = (__bf16)v_;
      }
}

// ---------------------------------------------------------------------------
// Host launch
// ---------------------------------------------------------------------------
extern "C" void kernel_launch(void* const* d_in, const int* in_sizes, int n_in,
                              void* d_out, int out_size, void* d_ws, size_t ws_size,
                              hipStream_t stream) {
  (void)in_sizes; (void)n_in; (void)out_size; (void)ws_size;
  const float* x   = (const float*)d_in[0];
  const float* ctx = (const float*)d_in[1];
  const float* Wq  = (const float*)d_in[2];
  const float* Wk  = (const float*)d_in[3];
  const float* Wv  = (const float*)d_in[4];
  const float* Wo  = (const float*)d_in[5];
  const float* bo  = (const float*)d_in[6];

  char* ws = (char*)d_ws;
  size_t off = 0;
  auto alloc_bf16 = [&](size_t elems) {
    __bf16* p = (__bf16*)(ws + off);
    off += ((elems * 2 + 255) / 256) * 256;
    return p;
  };
  __bf16* xb  = alloc_bf16((size_t)BATCH * NQ * DQ);
  __bf16* cb  = alloc_bf16((size_t)BATCH * NC * DC);
  __bf16* WqT = alloc_bf16((size_t)INNER * DQ);
  __bf16* WkT = alloc_bf16((size_t)INNER * DC);
  __bf16* WvT = alloc_bf16((size_t)INNER * DC);
  __bf16* WoT = alloc_bf16((size_t)DQ * INNER);
  __bf16* qp  = alloc_bf16((size_t)BATCH * NQ * INNER);
  __bf16* kp  = alloc_bf16((size_t)BATCH * NC * INNER);
  __bf16* vt  = alloc_bf16((size_t)BATCH * INNER * NC);
  __bf16* ao  = alloc_bf16((size_t)BATCH * NQ * INNER);

  const size_t nx = (size_t)BATCH * NQ * DQ;
  cvt_f32_bf16<<<(unsigned)((nx + 255) / 256), 256, 0, stream>>>(x, xb, nx);
  const size_t nctx = (size_t)BATCH * NC * DC;
  cvt_f32_bf16<<<(unsigned)((nctx + 255) / 256), 256, 0, stream>>>(ctx, cb, nctx);

  trans_f32_bf16<<<(unsigned)(((size_t)DQ * INNER + 255) / 256), 256, 0, stream>>>(Wq, WqT, DQ, INNER);
  trans_f32_bf16<<<(unsigned)(((size_t)DC * INNER + 255) / 256), 256, 0, stream>>>(Wk, WkT, DC, INNER);
  trans_f32_bf16<<<(unsigned)(((size_t)DC * INNER + 255) / 256), 256, 0, stream>>>(Wv, WvT, DC, INNER);
  trans_f32_bf16<<<(unsigned)(((size_t)INNER * DQ + 255) / 256), 256, 0, stream>>>(Wo, WoT, INNER, DQ);

  // q = (x @ Wq) * scale   -> qp [B*NQ][INNER] bf16
  gemm_bt_kernel<0><<<dim3(INNER / 128, BATCH * NQ / 128), 256, 0, stream>>>(
      xb, WqT, qp, BATCH * NQ, INNER, DQ, DQ, DQ, INNER, QSCALE, nullptr, 0);
  // k = context @ Wk       -> kp [B*NC][INNER] bf16
  gemm_bt_kernel<0><<<dim3(INNER / 128, BATCH * NC / 128), 256, 0, stream>>>(
      cb, WkT, kp, BATCH * NC, INNER, DC, DC, DC, INNER, 1.0f, nullptr, 0);
  // v = context @ Wv       -> vt [B][INNER][NC] bf16 (transposed per batch)
  gemm_bt_kernel<1><<<dim3(INNER / 128, BATCH * NC / 128), 256, 0, stream>>>(
      cb, WvT, vt, BATCH * NC, INNER, DC, DC, DC, 0, 1.0f, nullptr, NC);

  // attention -> ao [B*NQ][INNER] bf16
  flash_attn_kernel<<<dim3(NQ / 256, BATCH * HEADS), 256, 0, stream>>>(qp, kp, vt, ao);

  // out = ao @ Wo + bo -> d_out f32 [B*NQ][DQ]
  gemm_bt_kernel<2><<<dim3(DQ / 128, BATCH * NQ / 128), 256, 0, stream>>>(
      ao, WoT, d_out, BATCH * NQ, DQ, INNER, INNER, INNER, DQ, 1.0f, bo, 0);
}
